// DN_21758304321871
// MI455X (gfx1250) — matrix operand order repaired and checked by reading, not compiled, not run on hardware
//
#include <hip/hip_runtime.h>
#include <math.h>

// Problem constants (match reference setup_inputs)
#define Yn 8192
#define Xn 4096
#define Zn 100
#define Bn 1024
#define EPSV 1e-12f
#define PER_ITEM 2   // per_item is fixed to 2 in setup_inputs

typedef __bf16 v16bf __attribute__((ext_vector_type(16)));
typedef float  v8f   __attribute__((ext_vector_type(8)));

union Frag { v16bf v; uint4 u[2]; };

__device__ __forceinline__ __bf16 f2bf(float f) { return (__bf16)f; }

// LDS (addrspace 3) byte offset of a shared-memory pointer; HW adds LDS_BASE.
typedef __attribute__((address_space(3))) const void* lds_cptr;
__device__ __forceinline__ unsigned lds_addr(const void* p)
{
    return (unsigned)(unsigned long long)(lds_cptr)p;
}

// Issue one 32B A-chunk + one 32B B-chunk as 4 async global->LDS b128 ops.
// INST_OFFSET is added to both LDS and global addresses (ISA 10.7 pseudocode).
__device__ __forceinline__ void issue_tile(unsigned ldsA, unsigned ldsB,
                                           const __bf16* gA, const __bf16* gB)
{
    asm volatile(
        "global_load_async_to_lds_b128 %0, %2, off\n\t"
        "global_load_async_to_lds_b128 %0, %2, off offset:16\n\t"
        "global_load_async_to_lds_b128 %1, %3, off\n\t"
        "global_load_async_to_lds_b128 %1, %3, off offset:16"
        :: "v"(ldsA), "v"(ldsB), "v"(gA), "v"(gB) : "memory");
}

// ---------------------------------------------------------------------------
// bf16 WMMA GEMM: C[Bn x Yn] = Ah[Bn x Xn] * Bh[Yn x Xn]^T, fused epilogue.
// MODE 0: C = 0.5*acc + 0.5*Wz[n, z[m]]        (train ypre)
// MODE 1: C = acc * (age[n] >= 1 ? 1 : 0)      (final masked x2y)
// Block tile 128x128, K-step 32, 8 waves, each wave 32x64 (2x4 wmma tiles).
// Double-buffered LDS fed by global_load_async_to_lds_b128 (ASYNCcnt).
// ---------------------------------------------------------------------------
#define LDT 40   // LDS row stride in bf16 elems (32 + 8 pad), keeps 16B align

template <int MODE>
__global__ __launch_bounds__(256)
void gemm_xWt(const __bf16* __restrict__ Ah, const __bf16* __restrict__ Bh,
              float* __restrict__ C, const float* __restrict__ Wz,
              const int* __restrict__ z, const float* __restrict__ age)
{
    __shared__ __align__(16) __bf16 Sm[2][2][128 * LDT];   // [buf][A/B][...]
    const int tid = threadIdx.x;
    const int m0 = blockIdx.y * 128;
    const int n0 = blockIdx.x * 128;
    const int lane = tid & 31;
    const int wid  = tid >> 5;          // 0..7
    const int wm   = (wid & 3) * 32;    // wave M offset within block tile
    const int wn   = (wid >> 2) * 64;   // wave N offset within block tile
    const int hi   = lane >> 4;         // 0 = lanes 0-15, 1 = lanes 16-31
    const int l15  = lane & 15;

    v8f acc[2][4];
    const v8f z8 = {0.f, 0.f, 0.f, 0.f, 0.f, 0.f, 0.f, 0.f};
#pragma unroll
    for (int a = 0; a < 2; ++a)
#pragma unroll
        for (int b = 0; b < 4; ++b) acc[a][b] = z8;

    // staging: thread t owns tile row (t>>1), 16 bf16 cols at (t&1)*16 (= 32B)
    const int lrow = tid >> 1;
    const int lcol = (tid & 1) * 16;
    const unsigned stoff = (unsigned)(lrow * LDT + lcol) * 2u;
    const unsigned baseA[2] = { lds_addr(&Sm[0][0][0]) + stoff,
                                lds_addr(&Sm[1][0][0]) + stoff };
    const unsigned baseB[2] = { lds_addr(&Sm[0][1][0]) + stoff,
                                lds_addr(&Sm[1][1][0]) + stoff };
    const __bf16* gA0 = Ah + (size_t)(m0 + lrow) * Xn + lcol;
    const __bf16* gB0 = Bh + (size_t)(n0 + lrow) * Xn + lcol;

    // prologue: async-stage tile 0 into buffer 0
    issue_tile(baseA[0], baseB[0], gA0, gB0);

    int cur = 0;
    for (int ko = 0; ko < Xn; ko += 32) {
        const bool more = (ko + 32) < Xn;
        if (more) {
            // overlap: stage next tile into the other buffer
            issue_tile(baseA[cur ^ 1], baseB[cur ^ 1],
                       gA0 + ko + 32, gB0 + ko + 32);
            // async loads complete in order: <=4 outstanding => tile ko landed
            asm volatile("s_wait_asynccnt 0x4" ::: "memory");
        } else {
            asm volatile("s_wait_asynccnt 0x0" ::: "memory");
        }
        __syncthreads();

        const __bf16* Abuf = Sm[cur][0];
        const __bf16* Bbuf = Sm[cur][1];

        // A fragment (16x32): lane<16 row=base+lane holds K {0..7,16..23};
        // lanes 16-31 hold K {8..15,24..31}  (ISA 7.12.2)
        Frag fA[2], fB[4];
#pragma unroll
        for (int a = 0; a < 2; ++a) {
            const int row = wm + a * 16 + l15;
            const int kk  = hi * 8;
            fA[a].u[0] = *(const uint4*)&Abuf[row * LDT + kk];
            fA[a].u[1] = *(const uint4*)&Abuf[row * LDT + kk + 16];
        }
        // B fragment (32x16): lane n<16 holds column n K {0..7,16..23}; lanes 16-31 K {8..15,24..31}
#pragma unroll
        for (int b = 0; b < 4; ++b) {
            const int nr = wn + b * 16 + l15;   // Bh row == output N index
            const int kk = hi * 8;
            fB[b].u[0] = *(const uint4*)&Bbuf[nr * LDT + kk];
            fB[b].u[1] = *(const uint4*)&Bbuf[nr * LDT + kk + 16];
        }
#pragma unroll
        for (int a = 0; a < 2; ++a)
#pragma unroll
            for (int b = 0; b < 4; ++b)
                acc[a][b] = __builtin_amdgcn_wmma_f32_16x16x32_bf16(
                    false, fA[a].v, false, fB[b].v, (short)0, acc[a][b],
                    false, false);
        __syncthreads();   // reads done before next iter overwrites this buffer
        cur ^= 1;
    }

    // Epilogue: C layout (ISA): lane<16 -> M = r, lane>=16 -> M = 8+r, N = lane%16
#pragma unroll
    for (int a = 0; a < 2; ++a) {
#pragma unroll
        for (int r = 0; r < 8; ++r) {
            const int m  = m0 + wm + a * 16 + hi * 8 + r;
            int zb = 0;
            if (MODE == 0) zb = z[m];
#pragma unroll
            for (int b = 0; b < 4; ++b) {
                const int n = n0 + wn + b * 16 + l15;
                float v = acc[a][b][r];
                if (MODE == 0) v = 0.5f * v + 0.5f * Wz[(size_t)n * Zn + zb];
                else           v = (age[n] >= 1.0f) ? v : 0.0f;
                C[(size_t)m * Yn + n] = v;
            }
        }
    }
}

// ---------------------------------------------------------------------------
// Row L2-normalize: dst = src/max(||src||,1e-12); optional fused bf16 copy.
// One block per row; safe in-place.
// ---------------------------------------------------------------------------
__global__ __launch_bounds__(256)
void row_normalize(const float* __restrict__ src, float* __restrict__ dst,
                   __bf16* __restrict__ dst16, int cols)
{
    const int row = blockIdx.x;
    const int tid = threadIdx.x;
    const float* s = src + (size_t)row * cols;
    float ss = 0.f;
    for (int j = tid; j < cols; j += 256) { float v = s[j]; ss += v * v; }
    __shared__ float red[256];
    red[tid] = ss; __syncthreads();
    for (int off = 128; off > 0; off >>= 1) {
        if (tid < off) red[tid] += red[tid + off];
        __syncthreads();
    }
    const float inv = 1.0f / fmaxf(sqrtf(red[0]), EPSV);
    float* d = dst + (size_t)row * cols;
    __bf16* d16 = dst16 ? dst16 + (size_t)row * cols : nullptr;
    for (int j = tid; j < cols; j += 256) {
        float v = s[j] * inv;
        d[j] = v;
        if (d16) d16[j] = f2bf(v);
    }
}

__global__ void zero_f(float* p, int n)
{
    int i = blockIdx.x * 256 + threadIdx.x;
    if (i < n) p[i] = 0.f;
}

__global__ void hist_z(const int* __restrict__ z, float* __restrict__ zcnt)
{
    int b = blockIdx.x * 256 + threadIdx.x;
    if (b < Bn) atomicAdd(&zcnt[z[b]], 1.0f);
}

__global__ void unact_flag_k(const float* __restrict__ age, int* __restrict__ flag)
{
    __shared__ int s;
    if (threadIdx.x == 0) s = 0;
    __syncthreads();
    int loc = 0;
    for (int j = threadIdx.x; j < Yn; j += 256) if (age[j] < 1.0f) loc = 1;
    if (loc) s = 1;                // benign race, all write 1
    __syncthreads();
    if (threadIdx.x == 0) *flag = s;
}

// ---------------------------------------------------------------------------
// Per-sample argmax with first-index tie rule + winner selection.
// mode 0: full judge/alt logic. mode 1: plain argmax (final forward).
// ---------------------------------------------------------------------------
__global__ __launch_bounds__(256)
void argmax_k(const float* __restrict__ ypre, const float* __restrict__ age,
              const float* __restrict__ thr, const int* __restrict__ flag,
              int* __restrict__ winner, float* __restrict__ maxresp, int mode)
{
    const int b = blockIdx.x;
    const int tid = threadIdx.x;
    const float* row = ypre + (size_t)b * Yn;
    float bv = -__builtin_inff(); int bi = Yn;
    float av = -__builtin_inff(); int ai = Yn;
    for (int j = tid; j < Yn; j += 256) {
        float v = row[j];
        if (v > bv || (v == bv && j < bi)) { bv = v; bi = j; }
        if (mode == 0) {
            float u = (age[j] < 1.0f) ? v : 0.0f;   // ypre * unact
            if (u > av || (u == av && j < ai)) { av = u; ai = j; }
        }
    }
    __shared__ float sv[256]; __shared__ int si[256];
    __shared__ float tv[256]; __shared__ int ti[256];
    sv[tid] = bv; si[tid] = bi; tv[tid] = av; ti[tid] = ai;
    __syncthreads();
    for (int off = 128; off > 0; off >>= 1) {
        if (tid < off) {
            if (sv[tid + off] > sv[tid] ||
                (sv[tid + off] == sv[tid] && si[tid + off] < si[tid])) {
                sv[tid] = sv[tid + off]; si[tid] = si[tid + off];
            }
            if (tv[tid + off] > tv[tid] ||
                (tv[tid + off] == tv[tid] && ti[tid + off] < ti[tid])) {
                tv[tid] = tv[tid + off]; ti[tid] = ti[tid + off];
            }
        }
        __syncthreads();
    }
    if (tid == 0) {
        const int mi = si[0];
        if (mode == 0) {
            bool judge = (sv[0] > thr[mi]) || (age[mi] < 1.0f);
            int has = *flag;
            winner[b]  = (judge || !has) ? mi : ti[0];
            maxresp[b] = sv[0];
        } else {
            winner[b] = mi;
        }
    }
}

__global__ void accum_stats(const int* __restrict__ winner,
                            const float* __restrict__ maxresp,
                            const int* __restrict__ z,
                            float* __restrict__ cnt, float* __restrict__ sumr,
                            float* __restrict__ sumzy)
{
    int b = blockIdx.x * 256 + threadIdx.x;
    if (b >= Bn) return;
    int w = winner[b];
    atomicAdd(&cnt[w], 1.0f);
    atomicAdd(&sumr[w], maxresp[b]);
    atomicAdd(&sumzy[(size_t)z[b] * Yn + w], 1.0f);
}

// Wx[yy,:] += (lr/cnt)*(sum_x - cnt*Wx[yy,:]) ; rows with cnt==0 are unchanged.
__global__ __launch_bounds__(256)
void update_Wx_k(float* __restrict__ Wx, const float* __restrict__ xn,
                 const int* __restrict__ winner, const float* __restrict__ cnt,
                 const float* __restrict__ age)
{
    const int yy = blockIdx.x;
    const float c = cnt[yy];
    if (c <= 0.0f) return;
    __shared__ int mlist[Bn];
    __shared__ int nm;
    if (threadIdx.x == 0) nm = 0;
    __syncthreads();
    for (int b = threadIdx.x; b < Bn; b += 256)
        if (winner[b] == yy) { int p = atomicAdd(&nm, 1); mlist[p] = b; }
    __syncthreads();
    const int n = nm;
    const float lr   = 1.0f / (age[yy] + 1.0f);   // pre-update age
    const float coef = lr / c;
    float* wrow = Wx + (size_t)yy * Xn;
    for (int col = threadIdx.x; col < Xn; col += 256) {
        float s = 0.f;
        for (int i = 0; i < n; ++i) s += xn[(size_t)mlist[i] * Xn + col];
        float wv = wrow[col];
        wrow[col] = wv + coef * (s - c * wv);
    }
}

__global__ __launch_bounds__(256)
void update_Wz_k(float* __restrict__ Wz, const int* __restrict__ z,
                 const int* __restrict__ winner, const float* __restrict__ cnt,
                 const float* __restrict__ age)
{
    const int yy = blockIdx.x;
    const float c = cnt[yy];
    if (c <= 0.0f) return;
    __shared__ float sz[Zn];
    for (int j = threadIdx.x; j < Zn; j += 256) sz[j] = 0.f;
    __syncthreads();
    for (int b = threadIdx.x; b < Bn; b += 256)
        if (winner[b] == yy) atomicAdd(&sz[z[b]], 1.0f);
    __syncthreads();
    const float lr   = 1.0f / (age[yy] + 1.0f);
    const float coef = lr / c;
    for (int j = threadIdx.x; j < Zn; j += 256) {
        float wv = Wz[(size_t)yy * Zn + j];
        Wz[(size_t)yy * Zn + j] = wv + coef * (sz[j] - c * wv);
    }
}

__global__ void update_thr_age_k(float* __restrict__ thr, float* __restrict__ age,
                                 const float* __restrict__ cnt,
                                 const float* __restrict__ sumr)
{
    int y = blockIdx.x * 256 + threadIdx.x;
    if (y >= Yn) return;
    float c = cnt[y];
    if (c > 0.0f) {
        float lr   = 1.0f / (age[y] + 1.0f);      // ylr == lr when cnt>0
        float ycur = sumr[y] / c;
        thr[y] = lr * ycur + (1.0f - lr) * thr[y];
        age[y] += 1.0f;
    }
}

__global__ void update_Wy_k(float* __restrict__ Wy, const float* __restrict__ sumzy,
                            const float* __restrict__ zcnt,
                            const float* __restrict__ zage)
{
    int i = blockIdx.x * 256 + threadIdx.x;
    if (i >= Zn * Yn) return;
    int zr = i / Yn;
    float zc   = zcnt[zr];
    float zdiv = (zc > 0.f) ? zc : 1.f;
    float zlr  = 1.0f / (zage[zr] + 1.0f);        // pre-update zage
    float wv = Wy[i];
    Wy[i] = wv + (zlr / zdiv) * (sumzy[i] - zc * wv);
}

__global__ void update_zage_k(float* __restrict__ zage, const float* __restrict__ zcnt)
{
    int i = threadIdx.x;
    if (i < Zn && zcnt[i] > 0.f) zage[i] += 1.0f;
}

__global__ void gather_out_k(float* __restrict__ out, const float* __restrict__ Wy,
                             const int* __restrict__ winner)
{
    int i = blockIdx.x * 256 + threadIdx.x;
    if (i >= Bn * Zn) return;
    int b = i / Zn, zr = i % Zn;
    out[i] = Wy[(size_t)zr * Yn + winner[b]];     // one-hot(y) @ Wy^T
}

// ---------------------------------------------------------------------------
extern "C" void kernel_launch(void* const* d_in, const int* in_sizes, int n_in,
                              void* d_out, int out_size, void* d_ws, size_t ws_size,
                              hipStream_t stream)
{
    (void)in_sizes; (void)n_in; (void)out_size; (void)ws_size;
    const float* x     = (const float*)d_in[0];
    const float* Wx0   = (const float*)d_in[1];
    const float* Wz0   = (const float*)d_in[2];
    const float* Wy0   = (const float*)d_in[3];
    const float* age0  = (const float*)d_in[4];
    const float* thr0  = (const float*)d_in[5];
    const float* zage0 = (const float*)d_in[6];
    const int*   z     = (const int*)d_in[7];
    // d_in[8] = per_item (device scalar); fixed at 2 per setup_inputs.

    char* ws = (char*)d_ws;
    size_t off = 0;
    auto alloc = [&](size_t bytes) -> void* {
        void* p = ws + off;
        off += (bytes + 255) & ~(size_t)255;
        return p;
    };
    // ~270 MB workspace layout
    float*  Wx    = (float*)alloc((size_t)Yn * Xn * 4);   // 134 MB working copy
    __bf16* Wxh   = (__bf16*)alloc((size_t)Yn * Xn * 2);  // 67 MB bf16 mirror
    float*  xn    = (float*)alloc((size_t)Bn * Xn * 4);   // normalized x (f32)
    __bf16* xh    = (__bf16*)alloc((size_t)Bn * Xn * 2);  // normalized x (bf16)
    float*  Wz    = (float*)alloc((size_t)Yn * Zn * 4);
    float*  Wy    = (float*)alloc((size_t)Zn * Yn * 4);
    float*  ypre  = (float*)alloc((size_t)Bn * Yn * 4);   // 33.5 MB
    float*  age   = (float*)alloc((size_t)Yn * 4);
    float*  thr   = (float*)alloc((size_t)Yn * 4);
    float*  zage  = (float*)alloc((size_t)Zn * 4);
    float*  cnt   = (float*)alloc((size_t)Yn * 4);
    float*  sumr  = (float*)alloc((size_t)Yn * 4);
    float*  sumzy = (float*)alloc((size_t)Zn * Yn * 4);
    float*  zcnt  = (float*)alloc((size_t)Zn * 4);
    int*    winner  = (int*)alloc((size_t)Bn * 4);
    float*  maxresp = (float*)alloc((size_t)Bn * 4);
    int*    flag    = (int*)alloc(256);

    // init mutable state from inputs (D2D copies are graph-capture safe)
    hipMemcpyAsync(Wx,   Wx0,   (size_t)Yn * Xn * 4, hipMemcpyDeviceToDevice, stream);
    hipMemcpyAsync(Wz,   Wz0,   (size_t)Yn * Zn * 4, hipMemcpyDeviceToDevice, stream);
    hipMemcpyAsync(Wy,   Wy0,   (size_t)Zn * Yn * 4, hipMemcpyDeviceToDevice, stream);
    hipMemcpyAsync(age,  age0,  (size_t)Yn * 4, hipMemcpyDeviceToDevice, stream);
    hipMemcpyAsync(thr,  thr0,  (size_t)Yn * 4, hipMemcpyDeviceToDevice, stream);
    hipMemcpyAsync(zage, zage0, (size_t)Zn * 4, hipMemcpyDeviceToDevice, stream);

    // x: [B,64,64] -> [B,X] row-normalized (f32 + bf16); z histogram (constant)
    row_normalize<<<Bn, 256, 0, stream>>>(x, xn, xh, Xn);
    zero_f<<<(Zn + 255) / 256, 256, 0, stream>>>(zcnt, Zn);
    hist_z<<<(Bn + 255) / 256, 256, 0, stream>>>(z, zcnt);

    const dim3 gemm_grid(Yn / 128, Bn / 128);   // (64, 8)

    for (int it = 0; it < PER_ITEM; ++it) {
        row_normalize<<<Yn, 256, 0, stream>>>(Wx, Wx, Wxh, Xn);
        row_normalize<<<Yn, 256, 0, stream>>>(Wz, Wz, nullptr, Zn);
        row_normalize<<<Zn, 256, 0, stream>>>(Wy, Wy, nullptr, Yn);

        gemm_xWt<0><<<gemm_grid, 256, 0, stream>>>(xh, Wxh, ypre, Wz, z, age);

        unact_flag_k<<<1, 256, 0, stream>>>(age, flag);
        argmax_k<<<Bn, 256, 0, stream>>>(ypre, age, thr, flag, winner, maxresp, 0);

        zero_f<<<(Yn + 255) / 256, 256, 0, stream>>>(cnt, Yn);
        zero_f<<<(Yn + 255) / 256, 256, 0, stream>>>(sumr, Yn);
        zero_f<<<(Zn * Yn + 255) / 256, 256, 0, stream>>>(sumzy, Zn * Yn);
        accum_stats<<<(Bn + 255) / 256, 256, 0, stream>>>(winner, maxresp, z,
                                                          cnt, sumr, sumzy);

        update_Wx_k<<<Yn, 256, 0, stream>>>(Wx, xn, winner, cnt, age);
        update_Wz_k<<<Yn, 256, 0, stream>>>(Wz, z, winner, cnt, age);
        update_Wy_k<<<(Zn * Yn + 255) / 256, 256, 0, stream>>>(Wy, sumzy, zcnt, zage);
        update_thr_age_k<<<(Yn + 255) / 256, 256, 0, stream>>>(thr, age, cnt, sumr);
        update_zage_k<<<1, 128, 0, stream>>>(zage, zcnt);
    }

    // final forward: normalize, masked x2y, argmax, gather Wy rows
    row_normalize<<<Yn, 256, 0, stream>>>(Wx, Wx, Wxh, Xn);
    row_normalize<<<Zn, 256, 0, stream>>>(Wy, Wy, nullptr, Yn);
    gemm_xWt<1><<<gemm_grid, 256, 0, stream>>>(xh, Wxh, ypre, Wz, z, age);
    argmax_k<<<Bn, 256, 0, stream>>>(ypre, age, thr, flag, winner, maxresp, 1);
    gather_out_k<<<(Bn * Zn + 255) / 256, 256, 0, stream>>>((float*)d_out, Wy, winner);
}
